// Decoder_11888469475442
// MI455X (gfx1250) — matrix operand order repaired and checked by reading, not compiled, not run on hardware
//
// Point-Transformer encoder for MI455X (gfx1250, wave32, WMMA).
// All 1x1-conv GEMMs use a WMMA f16 (f32-accum) tiled kernel:
//   v_wmma_f32_16x16x32_f16, 64x128 output tile / 256-thread block (8 waves,
//   each wave computes a 2x2 grid of 16x16 tiles -> 4 WMMAs per K-chunk),
//   LDS-staged fp32->f16 tiles, branch-free K-clamped staging, next-chunk
//   global loads overlapped with current-chunk WMMAs.
// Call-site invariants: Co % 64 == 0, N % 128 == 0 (all GEMMs in this net).
// Irregular ops (FPS, KNN top-16, grouping, softmax-16, maxpool) are VALU kernels.
// Workspace arena (~360 MB assumed available in d_ws); attention MLP (64->256->64)
// is chunked over columns so the 256-wide intermediate never exceeds its region.
#include <hip/hip_runtime.h>

typedef _Float16 v16h __attribute__((ext_vector_type(16)));
typedef _Float16 v8h  __attribute__((ext_vector_type(8)));
typedef float    v8f  __attribute__((ext_vector_type(8)));

#define TILE_M 64
#define TILE_N 128
#define LDT    40   // LDS row stride (f16 elems): 32 + 8 pad, keeps 16B alignment

// ---------------------------------------------------------------------------
// GEMM: Y[b] = act( W(Co x Ci) * X[b](Ci x N, row ld = ldx) + bias [+ res] )
// grid = (N/128, Co/64, B); block = 256 (8 waves: 2 row-groups x 4 col-groups).
// ---------------------------------------------------------------------------
__global__ void gemm_f16_kernel(const float* __restrict__ W, const float* __restrict__ X,
                                const float* __restrict__ bias, const float* __restrict__ res,
                                float* __restrict__ Y, int Co, int Ci, int N, int ldx, int ldy,
                                long long sx, long long sy, long long sres, int relu)
{
    __shared__ __align__(16) _Float16 ldsW[TILE_M * LDT];
    __shared__ __align__(16) _Float16 ldsX[TILE_N * LDT];

    const int b   = blockIdx.z;
    const int gm0 = blockIdx.y * TILE_M;
    const int gn0 = blockIdx.x * TILE_N;
    const float* Xb = X + (size_t)b * sx;

    const int tid  = threadIdx.x;
    const int lane = tid & 31;
    const int wave = tid >> 5;
    const int wr   = wave >> 2;          // row group 0..1  -> rows wr*32 + 0..31
    const int wc   = wave & 3;           // col group 0..3  -> cols wc*32 + 0..31
    const int lr   = lane & 15;
    const int hi   = lane >> 4;

    v8f acc00 = {}, acc01 = {}, acc10 = {}, acc11 = {};

    float  vw[8];
    float4 vx[4];
    // K-guarded, branch-free staging (rows/cols never OOB: Co%64==0, N%128==0)
    auto stageW = [&](int kc) {
        #pragma unroll
        for (int p = 0; p < 8; ++p) {
            int t = tid + p * 256;
            int kk = t & 31, r = t >> 5;
            int gk = kc + kk;
            float v = W[(size_t)(gm0 + r) * Ci + (gk < Ci ? gk : 0)];
            vw[p] = (gk < Ci) ? v : 0.f;
        }
    };
    auto stageX = [&](int kc) {
        #pragma unroll
        for (int p = 0; p < 4; ++p) {
            int t = tid + p * 256;              // 0..1023
            int c4 = (t & 31) * 4;              // col in tile, 0..124 (16B aligned)
            int kk = t >> 5;                    // 0..31
            int gk = kc + kk;
            float4 v = *(const float4*)(Xb + (size_t)(gk < Ci ? gk : 0) * ldx + (gn0 + c4));
            if (gk >= Ci) v = make_float4(0.f, 0.f, 0.f, 0.f);
            vx[p] = v;
        }
    };
    auto commit = [&]() {
        #pragma unroll
        for (int p = 0; p < 8; ++p) {
            int t = tid + p * 256;
            ldsW[(t >> 5) * LDT + (t & 31)] = (_Float16)vw[p];
        }
        #pragma unroll
        for (int p = 0; p < 4; ++p) {
            int t = tid + p * 256;
            int c4 = (t & 31) * 4, kk = t >> 5;
            ldsX[(c4 + 0) * LDT + kk] = (_Float16)vx[p].x;
            ldsX[(c4 + 1) * LDT + kk] = (_Float16)vx[p].y;
            ldsX[(c4 + 2) * LDT + kk] = (_Float16)vx[p].z;
            ldsX[(c4 + 3) * LDT + kk] = (_Float16)vx[p].w;
        }
    };

    stageW(0); stageX(0);
    for (int kc = 0; kc < Ci; kc += 32) {
        commit();
        __syncthreads();
        int kn = kc + 32;
        if (kn < Ci) { stageW(kn); stageX(kn); }   // overlap with WMMA below

        // A frags (16x32 MxK): lane<16: row=lr, K 0..7/16..23; lane>=16: K 8..15/24..31
        union { v16h v; v8h h[2]; } A0, A1, B0, B1;
        const _Float16* a0 = &ldsW[(wr * 32 + lr) * LDT];
        const _Float16* a1 = a0 + 16 * LDT;
        A0.h[0] = *(const v8h*)(a0 + hi * 8);   A0.h[1] = *(const v8h*)(a0 + 16 + hi * 8);
        A1.h[0] = *(const v8h*)(a1 + hi * 8);   A1.h[1] = *(const v8h*)(a1 + 16 + hi * 8);
        // B frags (32x16 KxN): lane<16: col=lr K 0..7/16..23; lane>=16: K 8..15/24..31
        const _Float16* b0 = &ldsX[(wc * 32 + lr) * LDT];
        const _Float16* b1 = b0 + 16 * LDT;
        B0.h[0] = *(const v8h*)(b0 + hi * 8);   B0.h[1] = *(const v8h*)(b0 + 16 + hi * 8);
        B1.h[0] = *(const v8h*)(b1 + hi * 8);   B1.h[1] = *(const v8h*)(b1 + 16 + hi * 8);

        acc00 = __builtin_amdgcn_wmma_f32_16x16x32_f16(false, A0.v, false, B0.v, (short)0, acc00, false, false);
        acc01 = __builtin_amdgcn_wmma_f32_16x16x32_f16(false, A0.v, false, B1.v, (short)0, acc01, false, false);
        acc10 = __builtin_amdgcn_wmma_f32_16x16x32_f16(false, A1.v, false, B0.v, (short)0, acc10, false, false);
        acc11 = __builtin_amdgcn_wmma_f32_16x16x32_f16(false, A1.v, false, B1.v, (short)0, acc11, false, false);
        __syncthreads();
    }

    float* Yb = Y + (size_t)b * sy;
    const float* Rb = res ? res + (size_t)b * sres : nullptr;
    #pragma unroll
    for (int i = 0; i < 2; ++i) {
        #pragma unroll
        for (int j = 0; j < 2; ++j) {
            int col = gn0 + wc * 32 + j * 16 + lr;
            #pragma unroll
            for (int e = 0; e < 8; ++e) {      // C/D: vgpr e -> row e (+8 for hi lanes)
                int row = gm0 + wr * 32 + i * 16 + hi * 8 + e;
                float v = (i ? (j ? acc11[e] : acc10[e]) : (j ? acc01[e] : acc00[e])) + bias[row];
                if (Rb) v += Rb[(size_t)row * ldy + col];
                if (relu) v = fmaxf(v, 0.f);
                Yb[(size_t)row * ldy + col] = v;
            }
        }
    }
}

// ---------------------------------------------------------------------------
// Furthest point sampling: one block per batch. out[b,0]=0, then argmax scan.
// ---------------------------------------------------------------------------
__global__ void fps_kernel(const float* __restrict__ xyz, int N, int npoint, int* __restrict__ out)
{
    __shared__ float dist[2048];
    __shared__ float rv[256];
    __shared__ int   ri[256];
    __shared__ int   last;
    const int b = blockIdx.x;
    const float* px = xyz + (size_t)b * 3 * N;
    for (int i = threadIdx.x; i < N; i += 256) dist[i] = 1e10f;
    if (threadIdx.x == 0) { last = 0; out[(size_t)b * npoint] = 0; }
    __syncthreads();
    for (int j = 1; j < npoint; ++j) {
        float lx = px[last], ly = px[N + last], lz = px[2 * N + last];
        float best = -1.f; int bi = 0x7fffffff;
        for (int i = threadIdx.x; i < N; i += 256) {
            float dx = px[i] - lx, dy = px[N + i] - ly, dz = px[2 * N + i] - lz;
            float nd = fminf(dist[i], dx * dx + dy * dy + dz * dz);
            dist[i] = nd;
            if (nd > best) { best = nd; bi = i; }
        }
        rv[threadIdx.x] = best; ri[threadIdx.x] = bi;
        __syncthreads();
        for (int s = 128; s > 0; s >>= 1) {
            if (threadIdx.x < s) {
                float ov = rv[threadIdx.x + s]; int oi = ri[threadIdx.x + s];
                if (ov > rv[threadIdx.x] || (ov == rv[threadIdx.x] && oi < ri[threadIdx.x])) {
                    rv[threadIdx.x] = ov; ri[threadIdx.x] = oi;
                }
            }
            __syncthreads();
        }
        if (threadIdx.x == 0) { last = ri[0]; out[(size_t)b * npoint + j] = last; }
        __syncthreads();
    }
}

#define KNN_K 16
// one thread per query: insertion sort of 16 nearest among N refs
__global__ void knn_kernel(const float* __restrict__ ref, const float* __restrict__ qry,
                           int N, int M, int* __restrict__ idx, long long total)
{
    long long i = (long long)blockIdx.x * blockDim.x + threadIdx.x;
    if (i >= total) return;
    int b = (int)(i / M), m = (int)(i % M);
    const float* r = ref + (size_t)b * 3 * N;
    float qx = qry[((size_t)b * 3 + 0) * M + m];
    float qy = qry[((size_t)b * 3 + 1) * M + m];
    float qz = qry[((size_t)b * 3 + 2) * M + m];
    float bd[KNN_K]; int bj[KNN_K];
    #pragma unroll
    for (int k = 0; k < KNN_K; ++k) { bd[k] = 1e30f; bj[k] = 0; }
    for (int n = 0; n < N; ++n) {
        float dx = qx - r[n], dy = qy - r[N + n], dz = qz - r[2 * N + n];
        float d = dx * dx + dy * dy + dz * dz;
        if (d < bd[KNN_K - 1]) {
            int p = KNN_K - 1;
            while (p > 0 && bd[p - 1] > d) { bd[p] = bd[p - 1]; bj[p] = bj[p - 1]; --p; }
            bd[p] = d; bj[p] = n;
        }
    }
    #pragma unroll
    for (int k = 0; k < KNN_K; ++k) idx[i * KNN_K + k] = bj[k];
}

// out[b,c,m] = in[b,c,fidx[b,m]]
__global__ void gather_pts_kernel(const float* __restrict__ in, const int* __restrict__ fidx,
                                  float* __restrict__ out, int C, int N, int M, long long total)
{
    long long i = (long long)blockIdx.x * blockDim.x + threadIdx.x;
    if (i >= total) return;
    int b = (int)(i / ((long long)C * M));
    int r = (int)(i % ((long long)C * M));
    int c = r / M, m = r % M;
    out[i] = in[((size_t)b * C + c) * N + fidx[(size_t)b * M + m]];
}

// rows 0..2: sign*(xyz[c,idx]-nxyz[c,m]); rows 3..3+Cf-1: pts[c,idx]
__global__ void group_sub_kernel(const float* __restrict__ xyz, const float* __restrict__ nxyz,
                                 const float* __restrict__ pts, const int* __restrict__ idx,
                                 float* __restrict__ out, int Cf, int N, int M, int K,
                                 float sign, long long total)
{
    long long i = (long long)blockIdx.x * blockDim.x + threadIdx.x;
    if (i >= total) return;
    long long mk = (long long)M * K;
    int b = (int)(i / mk); long long col = i % mk; int m = (int)(col / K);
    int j = idx[(size_t)b * mk + col];
    int Ct = 3 + Cf;
    size_t ob = (size_t)b * Ct * mk + col;
    #pragma unroll
    for (int c = 0; c < 3; ++c)
        out[ob + (size_t)c * mk] =
            sign * (xyz[((size_t)b * 3 + c) * N + j] - nxyz[((size_t)b * 3 + c) * M + m]);
    for (int c = 0; c < Cf; ++c)
        out[ob + (size_t)(3 + c) * mk] = pts[((size_t)b * Cf + c) * N + j];
}

// out[b,c,m,k] = in[b,c,idx[b,m,k]]
__global__ void gather_mk_kernel(const float* __restrict__ in, const int* __restrict__ idx,
                                 float* __restrict__ out, int C, int N, int M, int K, long long total)
{
    long long i = (long long)blockIdx.x * blockDim.x + threadIdx.x;
    if (i >= total) return;
    long long mk = (long long)M * K;
    int b = (int)(i / mk); long long col = i % mk;
    int j = idx[i];
    for (int c = 0; c < C; ++c)
        out[((size_t)b * C + c) * mk + col] = in[((size_t)b * C + c) * N + j];
}

// out = q[b,c,m] - key + pe, all (B,C,M,K) flat except q (B,C,M)
__global__ void qkpe_kernel(const float* __restrict__ q, const float* __restrict__ key,
                            const float* __restrict__ pe, float* __restrict__ out,
                            int C, int M, int K, long long total)
{
    long long i = (long long)blockIdx.x * blockDim.x + threadIdx.x;
    if (i >= total) return;
    long long cmk = (long long)C * M * K;
    int b = (int)(i / cmk); long long r = i % cmk;
    int c = (int)(r / ((long long)M * K));
    long long col = r % ((long long)M * K);
    int m = (int)(col / K);
    out[i] = q[((size_t)b * C + c) * M + m] - key[i] + pe[i];
}

__global__ void softmax16_kernel(float* __restrict__ a, long long rows)
{
    long long i = (long long)blockIdx.x * blockDim.x + threadIdx.x;
    if (i >= rows) return;
    float* p = a + i * 16;
    float m = p[0];
    #pragma unroll
    for (int k = 1; k < 16; ++k) m = fmaxf(m, p[k]);
    float e[16]; float s = 0.f;
    #pragma unroll
    for (int k = 0; k < 16; ++k) { e[k] = expf(p[k] - m); s += e[k]; }
    float inv = 1.f / s;
    #pragma unroll
    for (int k = 0; k < 16; ++k) p[k] = e[k] * inv;
}

// agg[b,c,m] = sum_k attn * (value + pe)
__global__ void agg_kernel(const float* __restrict__ attn, const float* __restrict__ val,
                           const float* __restrict__ pe, float* __restrict__ out, long long rows)
{
    long long i = (long long)blockIdx.x * blockDim.x + threadIdx.x;
    if (i >= rows) return;
    float v = val[i]; float s = 0.f;
    #pragma unroll
    for (int k = 0; k < 16; ++k) s += attn[i * 16 + k] * (v + pe[i * 16 + k]);
    out[i] = s;
}

__global__ void maxpool_kernel(const float* __restrict__ in, float* __restrict__ out,
                               long long rows, int width)
{
    long long i = (long long)blockIdx.x * blockDim.x + threadIdx.x;
    if (i >= rows) return;
    const float* p = in + i * (long long)width;
    float m = p[0];
    for (int k = 1; k < width; ++k) m = fmaxf(m, p[k]);
    out[i] = m;
}

// out (B,3+Cf,M) = concat(xyz(B,3,M), pts(B,Cf,M))
__global__ void concat_kernel(const float* __restrict__ xyz, const float* __restrict__ pts,
                              float* __restrict__ out, int Cf, int M, long long total)
{
    long long i = (long long)blockIdx.x * blockDim.x + threadIdx.x;
    if (i >= total) return;
    int Ct = 3 + Cf;
    int b = (int)(i / ((long long)Ct * M));
    int r = (int)(i % ((long long)Ct * M));
    int c = r / M, m = r % M;
    out[i] = (c < 3) ? xyz[((size_t)b * 3 + c) * M + m]
                     : pts[((size_t)b * Cf + (c - 3)) * M + m];
}

struct TrP {
    const float *ls_w,*ls_b,*k_w,*k_b,*q_w,*q_b,*v_w,*v_b;
    const float *pm1_w,*pm1_b,*pm2_w,*pm2_b,*am1_w,*am1_b,*am2_w,*am2_b,*le_w,*le_b;
};

extern "C" void kernel_launch(void* const* d_in, const int* in_sizes, int n_in,
                              void* d_out, int out_size, void* d_ws, size_t ws_size,
                              hipStream_t stream)
{
    const int B = 32;
    auto F = [&](int i) { return (const float*)d_in[i]; };
    const float* pc = F(0);

    // Detect pytree flattening: dict insertion order vs sorted keys.
    bool ins = (in_sizes[5] == 64 * 128);   // insertion: tr1.ls_w ; sorted: sa2.w0(16768)
    int sa1_i, tr1_i, sa2_i, tr2_i, sa3_i;
    if (ins) { sa1_i = 1; tr1_i = 5;  sa2_i = 23; tr2_i = 27; sa3_i = 45; }
    else     { sa1_i = 1; sa2_i = 5;  sa3_i = 9;  tr1_i = 13; tr2_i = 31; }
    auto mkTr = [&](int t) {
        TrP p;
        if (ins) {
            p.ls_w=F(t+0);  p.ls_b=F(t+1);  p.k_w=F(t+2);   p.k_b=F(t+3);
            p.q_w=F(t+4);   p.q_b=F(t+5);   p.v_w=F(t+6);   p.v_b=F(t+7);
            p.pm1_w=F(t+8); p.pm1_b=F(t+9); p.pm2_w=F(t+10);p.pm2_b=F(t+11);
            p.am1_w=F(t+12);p.am1_b=F(t+13);p.am2_w=F(t+14);p.am2_b=F(t+15);
            p.le_w=F(t+16); p.le_b=F(t+17);
        } else {            // sorted: am1_b,am1_w,am2_b,am2_w,k_b,k_w,le_b,le_w,ls_b,ls_w,
                            //         pm1_b,pm1_w,pm2_b,pm2_w,q_b,q_w,v_b,v_w
            p.am1_b=F(t+0); p.am1_w=F(t+1); p.am2_b=F(t+2); p.am2_w=F(t+3);
            p.k_b=F(t+4);   p.k_w=F(t+5);   p.le_b=F(t+6);  p.le_w=F(t+7);
            p.ls_b=F(t+8);  p.ls_w=F(t+9);  p.pm1_b=F(t+10);p.pm1_w=F(t+11);
            p.pm2_b=F(t+12);p.pm2_w=F(t+13);p.q_b=F(t+14);  p.q_w=F(t+15);
            p.v_b=F(t+16);  p.v_w=F(t+17);
        }
        return p;
    };
    TrP tr1 = mkTr(tr1_i), tr2 = mkTr(tr2_i);

    // ---------------- workspace arena (floats) ----------------
    float* wsf = (float*)d_ws;
    size_t off = 0;
    auto take = [&](size_t n) { float* p = wsf + off; off += (n + 63) & ~(size_t)63; return p; };
    float* l1_xyz = take((size_t)3 * 512 * B);
    float* l1_pts = take((size_t)128 * 512 * B);
    float* l2_xyz = take((size_t)3 * 128 * B);
    float* l2_pts = take((size_t)256 * 128 * B);
    int*   fidx   = (int*)take((size_t)512 * B);
    int*   idx    = (int*)take((size_t)512 * 16 * B);
    float* T0 = take((size_t)64 * 512 * B);
    float* T1 = take((size_t)64 * 512 * B);
    float* T2 = take((size_t)64 * 512 * B);
    float* T3 = take((size_t)64 * 512 * B);
    float* RF = take((size_t)6  * 8192 * B);   // feats / pos_rel / feat3
    float* RA = take((size_t)64 * 8192 * B);   // h1 / key / attn
    float* RC = take((size_t)64 * 8192 * B);   // peh / attn_in / h2(sa2,sa3)
    float* RD = take((size_t)64 * 8192 * B);   // pe
    float* RB = take((size_t)128* 8192 * B);   // h2(sa1) / feat2 / am1-out chunks
    (void)ws_size; (void)n_in; (void)out_size; // arena assumed to fit d_ws

    auto G1 = [&](long long total) { return dim3((unsigned)((total + 255) / 256)); };
    auto GEMM = [&](const float* W, const float* X, const float* bi, const float* rs, float* Y,
                    int Co, int Ci, int N, int ldx, int ldy,
                    long long sx, long long sy, long long sr, int relu) {
        dim3 g((N + 127) / 128, (Co + 63) / 64, B);
        gemm_f16_kernel<<<g, 256, 0, stream>>>(W, X, bi, rs, Y, Co, Ci, N, ldx, ldy, sx, sy, sr, relu);
    };

    auto transformer = [&](const TrP& p, float* pts, const float* pos, int inch, int M) {
        const long long MK = (long long)M * 16;
        const int NK = (int)MK;
        // x = linear_start(pts)
        GEMM(p.ls_w, pts, p.ls_b, nullptr, T0, 64, inch, M, M, M, (long long)inch * M, 64LL * M, 0, 0);
        knn_kernel<<<G1((long long)B * M), 256, 0, stream>>>(pos, pos, M, M, idx, (long long)B * M);
        // key = group(conv_k(x))
        GEMM(p.k_w, T0, p.k_b, nullptr, T1, 64, 64, M, M, M, 64LL * M, 64LL * M, 0, 0);
        gather_mk_kernel<<<G1((long long)B * MK), 256, 0, stream>>>(T1, idx, RA, 64, M, M, 16, (long long)B * MK);
        GEMM(p.q_w, T0, p.q_b, nullptr, T2, 64, 64, M, M, M, 64LL * M, 64LL * M, 0, 0);
        GEMM(p.v_w, T0, p.v_b, nullptr, T3, 64, 64, M, M, M, 64LL * M, 64LL * M, 0, 0);
        // pos_rel = pos[c,m] - pos[c,idx]
        group_sub_kernel<<<G1((long long)B * MK), 256, 0, stream>>>(pos, pos, nullptr, idx, RF,
                                                                    0, M, M, 16, -1.f, (long long)B * MK);
        // pe = pm2(relu(pm1(pos_rel)))
        GEMM(p.pm1_w, RF, p.pm1_b, nullptr, RC, 64, 3, NK, NK, NK, 3 * MK, 64 * MK, 0, 1);
        GEMM(p.pm2_w, RC, p.pm2_b, nullptr, RD, 64, 64, NK, NK, NK, 64 * MK, 64 * MK, 0, 0);
        // attn_in = q - key + pe (overwrites peh region, key consumed here)
        qkpe_kernel<<<G1((long long)B * 64 * MK), 256, 0, stream>>>(T2, RA, RD, RC, 64, M, 16,
                                                                    (long long)B * 64 * MK);
        // attn = am2(relu(am1(attn_in))), chunked over columns (256-wide mid lives in RB)
        const int CH = (NK > 2048) ? 2048 : NK;
        for (int c0 = 0; c0 < NK; c0 += CH) {
            GEMM(p.am1_w, RC + c0, p.am1_b, nullptr, RB, 256, 64, CH, NK, CH,
                 64LL * NK, 256LL * CH, 0, 1);
            GEMM(p.am2_w, RB, p.am2_b, nullptr, RA + c0, 64, 256, CH, CH, NK,
                 256LL * CH, 64LL * NK, 0, 0);
        }
        softmax16_kernel<<<G1((long long)B * 64 * M), 256, 0, stream>>>(RA, (long long)B * 64 * M);
        agg_kernel<<<G1((long long)B * 64 * M), 256, 0, stream>>>(RA, T3, RD, T1, (long long)B * 64 * M);
        // pts = le(agg) + pts (in-place residual)
        GEMM(p.le_w, T1, p.le_b, pts, pts, inch, 64, M, M, M, 64LL * M,
             (long long)inch * M, (long long)inch * M, 0);
    };

    // ================= SA1: 2048 -> 512, k=16, MLP 6->64->128 =================
    fps_kernel<<<B, 256, 0, stream>>>(pc, 2048, 512, fidx);
    gather_pts_kernel<<<G1((long long)B * 3 * 512), 256, 0, stream>>>(pc, fidx, l1_xyz, 3, 2048, 512,
                                                                      (long long)B * 3 * 512);
    knn_kernel<<<G1((long long)B * 512), 256, 0, stream>>>(pc, l1_xyz, 2048, 512, idx, (long long)B * 512);
    group_sub_kernel<<<G1((long long)B * 8192), 256, 0, stream>>>(pc, l1_xyz, pc, idx, RF,
                                                                  3, 2048, 512, 16, 1.f, (long long)B * 8192);
    GEMM(F(sa1_i + 0), RF, F(sa1_i + 1), nullptr, RA, 64, 6, 8192, 8192, 8192,
         6LL * 8192, 64LL * 8192, 0, 1);
    GEMM(F(sa1_i + 2), RA, F(sa1_i + 3), nullptr, RB, 128, 64, 8192, 8192, 8192,
         64LL * 8192, 128LL * 8192, 0, 0);
    maxpool_kernel<<<G1((long long)B * 128 * 512), 256, 0, stream>>>(RB, l1_pts,
                                                                     (long long)B * 128 * 512, 16);
    // ================= TR1 =================
    transformer(tr1, l1_pts, l1_xyz, 128, 512);

    // ================= SA2: 512 -> 128, k=16, MLP 131->128->256 =================
    fps_kernel<<<B, 256, 0, stream>>>(l1_xyz, 512, 128, fidx);
    gather_pts_kernel<<<G1((long long)B * 3 * 128), 256, 0, stream>>>(l1_xyz, fidx, l2_xyz, 3, 512, 128,
                                                                      (long long)B * 3 * 128);
    knn_kernel<<<G1((long long)B * 128), 256, 0, stream>>>(l1_xyz, l2_xyz, 512, 128, idx, (long long)B * 128);
    group_sub_kernel<<<G1((long long)B * 2048), 256, 0, stream>>>(l1_xyz, l2_xyz, l1_pts, idx, RB,
                                                                  128, 512, 128, 16, 1.f, (long long)B * 2048);
    GEMM(F(sa2_i + 0), RB, F(sa2_i + 1), nullptr, RA, 128, 131, 2048, 2048, 2048,
         131LL * 2048, 128LL * 2048, 0, 1);
    GEMM(F(sa2_i + 2), RA, F(sa2_i + 3), nullptr, RC, 256, 128, 2048, 2048, 2048,
         128LL * 2048, 256LL * 2048, 0, 0);
    maxpool_kernel<<<G1((long long)B * 256 * 128), 256, 0, stream>>>(RC, l2_pts,
                                                                     (long long)B * 256 * 128, 16);
    // ================= TR2 =================
    transformer(tr2, l2_pts, l2_xyz, 256, 128);

    // ================= SA3: group-all, MLP 259->512->1024, max over n ==========
    concat_kernel<<<G1((long long)B * 259 * 128), 256, 0, stream>>>(l2_xyz, l2_pts, RF, 256, 128,
                                                                    (long long)B * 259 * 128);
    GEMM(F(sa3_i + 0), RF, F(sa3_i + 1), nullptr, RA, 512, 259, 128, 128, 128,
         259LL * 128, 512LL * 128, 0, 1);
    GEMM(F(sa3_i + 2), RA, F(sa3_i + 3), nullptr, RC, 1024, 512, 128, 128, 128,
         512LL * 128, 1024LL * 128, 0, 0);
    maxpool_kernel<<<G1((long long)B * 1024), 256, 0, stream>>>(RC, (float*)d_out,
                                                                (long long)B * 1024, 128);
}
